// SCNN_UNet_ConvLSTM_40218073759799
// MI455X (gfx1250) — compile-verified, matrix operand order checked by reading, not run on hardware
//
#include <hip/hip_runtime.h>
#include <hip/hip_bf16.h>
#include <math.h>

// ---------------------------------------------------------------------------
// Types for WMMA
// ---------------------------------------------------------------------------
typedef __attribute__((ext_vector_type(16))) _Float16 v16h;
typedef __attribute__((ext_vector_type(8)))  _Float16 v8h;
typedef __attribute__((ext_vector_type(8)))  float    v8f;

static __device__ __forceinline__ int imin(int a, int b) { return a < b ? a : b; }

// ---------------------------------------------------------------------------
// Weight repack: (Cout, Cin, 3, 3) f32  ->  (Cout, 9, Cpad) f16, zero padded.
// K-order inside the GEMM is (tap, cin).
// ---------------------------------------------------------------------------
__global__ void k_pack_w(const float* __restrict__ w, _Float16* __restrict__ wp,
                         int Cout, int Cin, int Cpad)
{
  const long total = (long)Cout * 9 * Cpad;
  long i = (long)blockIdx.x * blockDim.x + threadIdx.x;
  if (i >= total) return;
  const int c   = (int)(i % Cpad);
  long t = i / Cpad;
  const int tap = (int)(t % 9);
  const int co  = (int)(t / 9);
  float v = 0.f;
  if (c < Cin) v = w[((long)co * Cin + c) * 9 + tap];
  wp[i] = (_Float16)v;
}

// ---------------------------------------------------------------------------
// Activation convert: (N, C, H, W) f32 -> (N, H, W, Cpad) f16, zero padded.
// ---------------------------------------------------------------------------
__global__ void k_nhwc_h(const float* __restrict__ in, _Float16* __restrict__ out,
                         int N, int C, int Cpad, int HW)
{
  const long total = (long)N * HW * Cpad;
  long i = (long)blockIdx.x * blockDim.x + threadIdx.x;
  if (i >= total) return;
  const int c   = (int)(i % Cpad);
  long t = i / Cpad;
  const int pix = (int)(t % HW);
  const int n   = (int)(t / HW);
  float v = 0.f;
  if (c < C) v = in[((long)n * C + c) * HW + pix];
  out[i] = (_Float16)v;
}

// ---------------------------------------------------------------------------
// Generic 3x3 SAME conv, stride 1, implicit GEMM via v_wmma_f32_16x16x32_f16.
//   ah  : (N, H, W, Cpad) f16 activations (zero padded channels)
//   wp  : (Cout, 9, Cpad) f16 packed weights
//   zpix: Cpad zeros (f16) -- border taps / out-of-range rows point here
//   out : (N, Cout, H, W) f32
// Each wave owns a 16(M) x 64(Cout) tile: one shared A fragment feeds 4 WMMAs
// per (tap, cin-block) step -> 2.5 b128 loads per wmma instead of 4.
// fuse==1: y = relu((conv+bias)*bn_scale + bn_shift); fuse==0: y = conv+bias.
// ---------------------------------------------------------------------------
__global__ void __launch_bounds__(256)
k_conv3x3_wmma(const _Float16* __restrict__ ah, const _Float16* __restrict__ wp,
               const _Float16* __restrict__ zpix,
               const float* __restrict__ bias,
               const float* __restrict__ bng, const float* __restrict__ bnb,
               const float* __restrict__ bnm, const float* __restrict__ bnv,
               float* __restrict__ out,
               int N, int H, int W, int Cout, int Cpad, int fuse)
{
  const int HW   = H * W;
  const int M    = N * HW;
  const int nMt  = (M + 15) >> 4;
  const int nNt4 = Cout >> 6;             // 64-wide output-channel blocks

  const int wave = threadIdx.x >> 5;
  const int tile = blockIdx.x * 8 + wave;
  if (tile >= nMt * nNt4) return;         // uniform per wave

  const int mt    = tile / nNt4;
  const int nt4   = tile % nNt4;
  const int lane  = threadIdx.x & 31;
  const int lrow  = lane & 15;
  const int l8    = (lane >> 4) << 3;     // 0 or 8 (halves)

  // A-row coordinates for this lane (M = flattened (n,y,x))
  const int  m    = mt * 16 + lrow;
  const bool mOK  = (m < M);
  const int  mm   = mOK ? m : 0;
  const int  nimg = mm / HW;
  const int  rem  = mm % HW;
  const int  ay   = rem / W;
  const int  ax   = rem % W;

  // Per-tap A base pointers (clamped to the zero pixel when out of bounds).
  const _Float16* abase[9];
#pragma unroll
  for (int tap = 0; tap < 9; ++tap) {
    const int dy = tap / 3 - 1;
    const int dx = tap % 3 - 1;
    const int yy = ay + dy;
    const int xx = ax + dx;
    const bool ok = mOK && (unsigned)yy < (unsigned)H && (unsigned)xx < (unsigned)W;
    abase[tap] = ok ? (ah + (((long)nimg * H + yy) * W + xx) * Cpad) : zpix;
  }

  // B bases for this lane's 4 output-channel sub-tiles
  const _Float16* wbase[4];
#pragma unroll
  for (int j = 0; j < 4; ++j)
    wbase[j] = wp + (long)(nt4 * 64 + 16 * j + lrow) * 9 * Cpad;

  v8f acc[4];
#pragma unroll
  for (int j = 0; j < 4; ++j)
    acc[j] = (v8f){0.f, 0.f, 0.f, 0.f, 0.f, 0.f, 0.f, 0.f};

  for (int c0 = 0; c0 < Cpad; c0 += 32) {
#pragma unroll
    for (int tap = 0; tap < 9; ++tap) {
      const _Float16* ap = abase[tap] + c0 + l8;
      const v8h a0 = *(const v8h*)ap;
      const v8h a1 = *(const v8h*)(ap + 16);
      v16h a;
#pragma unroll
      for (int i = 0; i < 8; ++i) { a[i] = a0[i]; a[i + 8] = a1[i]; }

#pragma unroll
      for (int j = 0; j < 4; ++j) {
        const _Float16* bp = wbase[j] + (long)tap * Cpad + c0 + l8;
        const v8h b0 = *(const v8h*)bp;
        const v8h b1 = *(const v8h*)(bp + 16);
        v16h b;
#pragma unroll
        for (int i = 0; i < 8; ++i) { b[i] = b0[i]; b[i + 8] = b1[i]; }
        acc[j] = __builtin_amdgcn_wmma_f32_16x16x32_f16(
            false, a, false, b, (short)0, acc[j], false, false);
      }
    }
  }

  // Epilogue: bias (+ fused BN + ReLU) for each of the 4 channel sub-tiles.
#pragma unroll
  for (int j = 0; j < 4; ++j) {
    const int co = nt4 * 64 + 16 * j + lrow;
    float sc = 1.f, sh = 0.f;
    if (fuse) {
      const float inv = rsqrtf(bnv[co] + 1e-5f);
      sc = bng[co] * inv;
      sh = bnb[co] - bng[co] * bnm[co] * inv;
    }
    const float bb = bias[co];
#pragma unroll
    for (int r = 0; r < 8; ++r) {
      const int mrow = mt * 16 + r + l8;  // l8 == 8*(lane>>4): C/D row offset
      if (mrow < M) {
        const int ni = mrow / HW;
        const int rr = mrow % HW;
        float v = acc[j][r] + bb;
        if (fuse) { v = v * sc + sh; v = v > 0.f ? v : 0.f; }
        out[((long)ni * Cout + co) * HW + rr] = v;
      }
    }
  }
}

// ---------------------------------------------------------------------------
// SCNN directional scan (sequential): one block per batch image.
// buf: (B, C, H, W) in-place.  Step s: row[pos] += relu(conv1d(row[prev])).
// ---------------------------------------------------------------------------
__global__ void __launch_bounds__(1024)
k_scnn_scan(float* __restrict__ buf, const float* __restrict__ w,
            int C, int S, int L, long sB, long sC, long sPos, long sJ,
            int reverse)
{
  float* base = buf + (long)blockIdx.x * sB;
  for (int s = 1; s < S; ++s) {
    const int pos  = reverse ? (S - 1 - s) : s;
    const int prev = reverse ? (pos + 1) : (pos - 1);
    for (int idx = threadIdx.x; idx < C * L; idx += blockDim.x) {
      const int o  = idx % L;
      const int co = idx / L;
      const float* __restrict__ wr = w + (long)co * C * 9;
      float acc = 0.f;
      for (int ci = 0; ci < C; ++ci) {
        const float* __restrict__ prow = base + (long)ci * sC + (long)prev * sPos;
        const float* __restrict__ wt   = wr + ci * 9;
#pragma unroll
        for (int t = 0; t < 9; ++t) {
          const int j = o + t - 4;
          if ((unsigned)j < (unsigned)L) acc += wt[t] * prow[(long)j * sJ];
        }
      }
      float* op = base + (long)co * sC + (long)pos * sPos + (long)o * sJ;
      *op += (acc > 0.f ? acc : 0.f);
    }
    __threadfence();
    __syncthreads();
  }
}

__global__ void k_maxpool2(const float* __restrict__ in, float* __restrict__ out,
                           int NC, int H, int W)
{
  const int Ho = H >> 1, Wo = W >> 1;
  const long total = (long)NC * Ho * Wo;
  long i = (long)blockIdx.x * blockDim.x + threadIdx.x;
  if (i >= total) return;
  const int xo = (int)(i % Wo);
  long t = i / Wo;
  const int yo = (int)(t % Ho);
  const int nc = (int)(t / Ho);
  const float* p = in + ((long)nc * H + yo * 2) * W + xo * 2;
  const float a = p[0] > p[1] ? p[0] : p[1];
  const float b = p[W] > p[W + 1] ? p[W] : p[W + 1];
  out[i] = a > b ? a : b;
}

__global__ void k_up2x(const float* __restrict__ in, float* __restrict__ out,
                       int NC, int H, int W)
{
  const int Ho = 2 * H, Wo = 2 * W;
  const long total = (long)NC * Ho * Wo;
  long i = (long)blockIdx.x * blockDim.x + threadIdx.x;
  if (i >= total) return;
  const int x = (int)(i % Wo);
  long t = i / Wo;
  const int y  = (int)(t % Ho);
  const int nc = (int)(t / Ho);

  const float py = (float)y * (float)(H - 1) / (float)(2 * H - 1);
  int  ly = (int)floorf(py);
  int  hy = imin(ly + 1, H - 1);
  const float fy = py - (float)ly;

  const float px = (float)x * (float)(W - 1) / (float)(2 * W - 1);
  int  lx = (int)floorf(px);
  int  hx = imin(lx + 1, W - 1);
  const float fx = px - (float)lx;

  const float* p = in + (long)nc * H * W;
  const float t0 = p[(long)ly * W + lx] * (1.f - fy) + p[(long)hy * W + lx] * fy;
  const float t1 = p[(long)ly * W + hx] * (1.f - fy) + p[(long)hy * W + hx] * fy;
  out[i] = t0 * (1.f - fx) + t1 * fx;
}

__global__ void k_lstm_gates(const float* __restrict__ z,
                             float* __restrict__ c, float* __restrict__ h,
                             float* __restrict__ hs, int B, int Cg, int HW)
{
  const long total = (long)B * Cg * HW;
  long i = (long)blockIdx.x * blockDim.x + threadIdx.x;
  if (i >= total) return;
  const int hw = (int)(i % HW);
  long t = i / HW;
  const int ch = (int)(t % Cg);
  const int b  = (int)(t / Cg);
  const float* zb = z + (long)b * 4 * Cg * HW;
  const float zi = zb[((long)0 * Cg + ch) * HW + hw];
  const float zf = zb[((long)1 * Cg + ch) * HW + hw];
  const float zo = zb[((long)2 * Cg + ch) * HW + hw];
  const float zg = zb[((long)3 * Cg + ch) * HW + hw];
  const float si = 1.f / (1.f + expf(-zi));
  const float sf = 1.f / (1.f + expf(-zf));
  const float so = 1.f / (1.f + expf(-zo));
  const float cn = sf * c[i] + si * tanhf(zg);
  const float hn = so * tanhf(cn);
  c[i] = cn; h[i] = hn; hs[i] = hn;
}

__global__ void k_conv1x1(const float* __restrict__ in, const float* __restrict__ w,
                          const float* __restrict__ b, float* __restrict__ out,
                          int N, int Cin, int Cout, int HW)
{
  const long total = (long)N * Cout * HW;
  long i = (long)blockIdx.x * blockDim.x + threadIdx.x;
  if (i >= total) return;
  const int hw = (int)(i % HW);
  long t = i / HW;
  const int co = (int)(t % Cout);
  const int n  = (int)(t / Cout);
  const float* ip = in + (long)n * Cin * HW + hw;
  float s = b[co];
  for (int ci = 0; ci < Cin; ++ci) s += w[co * Cin + ci] * ip[(long)ci * HW];
  out[i] = s;
}

__global__ void k_copy_chan(const float* __restrict__ src, float* __restrict__ dst,
                            int B, long chanElems, long srcBS, long srcOff,
                            long dstBS, long dstOff)
{
  const long total = (long)B * chanElems;
  long i = (long)blockIdx.x * blockDim.x + threadIdx.x;
  if (i >= total) return;
  const long b = i / chanElems;
  const long r = i % chanElems;
  dst[b * dstBS + dstOff + r] = src[b * srcBS + srcOff + r];
}

__global__ void k_zero(float* __restrict__ p, long n)
{
  long i = (long)blockIdx.x * blockDim.x + threadIdx.x;
  if (i < n) p[i] = 0.f;
}

__global__ void k_zero_h(_Float16* __restrict__ p, long n)
{
  long i = (long)blockIdx.x * blockDim.x + threadIdx.x;
  if (i < n) p[i] = (_Float16)0.f;
}

// ---------------------------------------------------------------------------
// Host-side orchestration
// ---------------------------------------------------------------------------
static inline int nblk(long n) { return (int)((n + 255) / 256); }

// Input layout: jax tree_leaves (dicts flattened by sorted key) of
// {'params': {...}, 'x': x}.  Per double-conv dict (12 leaves):
//   +0 b1, +1 b2, +2 bn1.beta, +3 bn1.gamma, +4 bn1.mean, +5 bn1.var,
//   +6 bn2.beta, +7 bn2.gamma, +8 bn2.mean, +9 bn2.var, +10 w1, +11 w2
enum {
  DOWN1 = 0, DOWN2 = 12, DOWN3 = 24, DOWN4 = 36, INC = 48,
  LSTM0_B = 60, LSTM0_W = 61, LSTM1_B = 62, LSTM1_W = 63,
  OUTC_B = 64, OUTC_W = 65,
  SCNN_D = 66, SCNN_L = 67, SCNN_R = 68, SCNN_U = 69,
  UP1 = 70, UP2 = 82, UP3 = 94, UP4 = 106,
  X_IN = 118
};

struct ConvCtx {
  _Float16* ah;    // NHWC f16 activation staging
  _Float16* wpk;   // packed f16 weight staging
  _Float16* zpix;  // zeroed pixel (maxCpad halves)
};

static void conv3(hipStream_t s, const ConvCtx& cx,
                  const float* in, const float* w, const float* b,
                  const float* g, const float* be, const float* mn, const float* vr,
                  float* out, int N, int Cin, int H, int W, int Cout, int fuse)
{
  const int Cpad = (Cin + 31) & ~31;
  const long pw = (long)Cout * 9 * Cpad;
  k_pack_w<<<nblk(pw), 256, 0, s>>>(w, cx.wpk, Cout, Cin, Cpad);
  const long pa = (long)N * H * W * Cpad;
  k_nhwc_h<<<nblk(pa), 256, 0, s>>>(in, cx.ah, N, Cin, Cpad, H * W);
  const long tiles = (long)((N * H * W + 15) / 16) * (Cout / 64);
  const int blocks = (int)((tiles + 7) / 8);
  k_conv3x3_wmma<<<blocks, 256, 0, s>>>(cx.ah, cx.wpk, cx.zpix,
                                        b, g, be, mn, vr, out,
                                        N, H, W, Cout, Cpad, fuse);
}

static void dconv(hipStream_t s, const ConvCtx& cx, void* const* d_in, int base,
                  const float* in, float* tmp, float* out,
                  int N, int Cin, int H, int W, int Cout)
{
  const float* P[12];
  for (int i = 0; i < 12; ++i) P[i] = (const float*)d_in[base + i];
  conv3(s, cx, in,  P[10], P[0], P[3], P[2], P[4], P[5], tmp, N, Cin,  H, W, Cout, 1);
  conv3(s, cx, tmp, P[11], P[1], P[7], P[6], P[8], P[9], out, N, Cout, H, W, Cout, 1);
}

extern "C" void kernel_launch(void* const* d_in, const int* in_sizes, int n_in,
                              void* d_out, int out_size, void* d_ws, size_t ws_size,
                              hipStream_t stream)
{
  (void)in_sizes; (void)n_in; (void)out_size; (void)ws_size;
  const int B = 2, T = 5, H0 = 128, W0 = 256;
  const long HW0 = (long)H0 * W0;                // 32768

  const float* X = (const float*)d_in[X_IN];     // (2,5,3,128,256)
  float* OUT = (float*)d_out;                    // 131072 (out) + 131072 (test)

  // ---- workspace carving ----
  char* wp = (char*)d_ws;
  auto allocB = [&](long bytes) -> char* {
    char* r = wp;
    wp += ((bytes + 255) / 256) * 256;
    return r;
  };
  auto alloc  = [&](long elems) -> float*    { return (float*)allocB(elems * 4); };
  auto allocH = [&](long elems) -> _Float16* { return (_Float16*)allocB(elems * 2); };

  float* xin = alloc((long)B * 3 * HW0);
  float* x1  = alloc((long)B * 64 * HW0);
  float* t1  = alloc((long)B * 64 * HW0);
  float* p1  = alloc((long)B * 64 * (HW0 / 4));
  float* x2  = alloc((long)B * 128 * (HW0 / 4));
  float* t2  = alloc((long)B * 128 * (HW0 / 4));
  float* p2  = alloc((long)B * 128 * (HW0 / 16));
  float* x3  = alloc((long)B * 256 * (HW0 / 16));
  float* t3  = alloc((long)B * 256 * (HW0 / 16));
  float* p3  = alloc((long)B * 256 * (HW0 / 64));
  float* x4  = alloc((long)B * 512 * (HW0 / 64));
  float* t4  = alloc((long)B * 512 * (HW0 / 64));
  float* p4  = alloc((long)B * 512 * (HW0 / 256));
  float* t5  = alloc((long)B * 512 * (HW0 / 256));
  float* x5s = alloc((long)T * B * 512 * (HW0 / 256));
  float* hb  = alloc((long)B * 512 * (HW0 / 256));
  float* cb  = alloc((long)B * 512 * (HW0 / 256));
  float* cat = alloc((long)B * 1024 * (HW0 / 256));
  float* zbuf= alloc((long)B * 2048 * (HW0 / 256));
  float* hs1 = alloc((long)T * B * 512 * (HW0 / 256));
  float* hs2 = alloc((long)T * B * 512 * (HW0 / 256));
  float* upb = alloc((long)B * 64 * HW0);
  float* c1b = alloc((long)B * 1024 * (HW0 / 64));
  float* y1  = alloc((long)B * 256 * (HW0 / 64));
  float* c2b = alloc((long)B * 512 * (HW0 / 16));
  float* y2  = alloc((long)B * 128 * (HW0 / 16));
  float* c3b = alloc((long)B * 256 * (HW0 / 4));
  float* y3  = alloc((long)B * 64 * (HW0 / 4));
  float* c4b = alloc((long)B * 128 * HW0);
  float* y4  = alloc((long)B * 64 * HW0);
  float* dt  = alloc((long)B * 64 * HW0);

  ConvCtx cx;
  cx.ah   = allocH((long)B * HW0 * 128);           // max NHWC f16: cat4 conv1
  cx.wpk  = allocH((long)2048 * 9 * 1024);         // max packed weights: LSTM
  cx.zpix = allocH(1024);
  k_zero_h<<<nblk(1024), 256, 0, stream>>>(cx.zpix, 1024);

  const float* wD = (const float*)d_in[SCNN_D];
  const float* wU = (const float*)d_in[SCNN_U];
  const float* wR = (const float*)d_in[SCNN_R];
  const float* wL = (const float*)d_in[SCNN_L];

  // ================= Encoder: all 5 frames (skips keep last frame) ========
  for (int t = 0; t < T; ++t) {
    {
      long ce = 3 * HW0;
      k_copy_chan<<<nblk((long)B * ce), 256, 0, stream>>>(
          X, xin, B, ce, (long)T * 3 * HW0, (long)t * 3 * HW0, 3 * HW0, 0);
    }
    dconv(stream, cx, d_in, INC, xin, t1, x1, B, 3, H0, W0, 64);

    // scnn on a copy of x1 (x1 itself is the skip connection)
    {
      long ce = 64 * HW0;
      k_copy_chan<<<nblk((long)B * ce), 256, 0, stream>>>(
          x1, t1, B, ce, ce, 0, ce, 0);
      const long sB = 64 * HW0, sC = HW0;
      k_scnn_scan<<<B, 1024, 0, stream>>>(t1, wD, 64, H0, W0, sB, sC, W0, 1, 0);
      k_scnn_scan<<<B, 1024, 0, stream>>>(t1, wU, 64, H0, W0, sB, sC, W0, 1, 1);
      k_scnn_scan<<<B, 1024, 0, stream>>>(t1, wR, 64, W0, H0, sB, sC, 1, W0, 0);
      k_scnn_scan<<<B, 1024, 0, stream>>>(t1, wL, 64, W0, H0, sB, sC, 1, W0, 1);
    }
    k_maxpool2<<<nblk((long)B * 64 * HW0 / 4), 256, 0, stream>>>(t1, p1, B * 64, H0, W0);
    dconv(stream, cx, d_in, DOWN1, p1, t2, x2, B, 64, 64, 128, 128);
    k_maxpool2<<<nblk((long)B * 128 * HW0 / 16), 256, 0, stream>>>(x2, p2, B * 128, 64, 128);
    dconv(stream, cx, d_in, DOWN2, p2, t3, x3, B, 128, 32, 64, 256);
    k_maxpool2<<<nblk((long)B * 256 * HW0 / 64), 256, 0, stream>>>(x3, p3, B * 256, 32, 64);
    dconv(stream, cx, d_in, DOWN3, p3, t4, x4, B, 256, 16, 32, 512);
    k_maxpool2<<<nblk((long)B * 512 * HW0 / 256), 256, 0, stream>>>(x4, p4, B * 512, 16, 32);
    dconv(stream, cx, d_in, DOWN4, p4, t5, x5s + (long)t * B * 512 * 128, B, 512, 8, 16, 512);
  }

  // ================= ConvLSTM stack (2 layers, hidden 512, 8x16) ==========
  const long fst = (long)B * 512 * 128;  // per-timestep frame elems (131072)
  for (int layer = 0; layer < 2; ++layer) {
    const float* lw = (const float*)d_in[layer == 0 ? LSTM0_W : LSTM1_W];
    const float* lb = (const float*)d_in[layer == 0 ? LSTM0_B : LSTM1_B];
    const float* in_seq = (layer == 0) ? x5s : hs1;
    float* out_seq = (layer == 0) ? hs1 : hs2;
    k_zero<<<nblk(fst), 256, 0, stream>>>(hb, fst);
    k_zero<<<nblk(fst), 256, 0, stream>>>(cb, fst);
    for (int t = 0; t < T; ++t) {
      const long ce = 512 * 128;
      k_copy_chan<<<nblk((long)B * ce), 256, 0, stream>>>(
          in_seq + (long)t * fst, cat, B, ce, ce, 0, 2 * ce, 0);
      k_copy_chan<<<nblk((long)B * ce), 256, 0, stream>>>(
          hb, cat, B, ce, ce, 0, 2 * ce, ce);
      conv3(stream, cx, cat, lw, lb, lb, lb, lb, lb, zbuf, B, 1024, 8, 16, 2048, 0);
      k_lstm_gates<<<nblk(fst), 256, 0, stream>>>(
          zbuf, cb, hb, out_seq + (long)t * fst, B, 512, 128);
    }
  }
  const float* test = hs2 + (long)(T - 1) * fst;  // (2,512,8,16)

  // ================= Decoder =============================================
  k_up2x<<<nblk((long)B * 512 * 512), 256, 0, stream>>>(test, upb, B * 512, 8, 16);
  {
    const long ce = 512 * 512;
    k_copy_chan<<<nblk((long)B * ce), 256, 0, stream>>>(x4,  c1b, B, ce, ce, 0, 2 * ce, 0);
    k_copy_chan<<<nblk((long)B * ce), 256, 0, stream>>>(upb, c1b, B, ce, ce, 0, 2 * ce, ce);
  }
  dconv(stream, cx, d_in, UP1, c1b, dt, y1, B, 1024, 16, 32, 256);

  k_up2x<<<nblk((long)B * 256 * 2048), 256, 0, stream>>>(y1, upb, B * 256, 16, 32);
  {
    const long ce = 256 * 2048;
    k_copy_chan<<<nblk((long)B * ce), 256, 0, stream>>>(x3,  c2b, B, ce, ce, 0, 2 * ce, 0);
    k_copy_chan<<<nblk((long)B * ce), 256, 0, stream>>>(upb, c2b, B, ce, ce, 0, 2 * ce, ce);
  }
  dconv(stream, cx, d_in, UP2, c2b, dt, y2, B, 512, 32, 64, 128);

  k_up2x<<<nblk((long)B * 128 * 8192), 256, 0, stream>>>(y2, upb, B * 128, 32, 64);
  {
    const long ce = 128 * 8192;
    k_copy_chan<<<nblk((long)B * ce), 256, 0, stream>>>(x2,  c3b, B, ce, ce, 0, 2 * ce, 0);
    k_copy_chan<<<nblk((long)B * ce), 256, 0, stream>>>(upb, c3b, B, ce, ce, 0, 2 * ce, ce);
  }
  dconv(stream, cx, d_in, UP3, c3b, dt, y3, B, 256, 64, 128, 64);

  k_up2x<<<nblk((long)B * 64 * HW0), 256, 0, stream>>>(y3, upb, B * 64, 64, 128);
  {
    const long ce = 64 * HW0;
    k_copy_chan<<<nblk((long)B * ce), 256, 0, stream>>>(x1,  c4b, B, ce, ce, 0, 2 * ce, 0);
    k_copy_chan<<<nblk((long)B * ce), 256, 0, stream>>>(upb, c4b, B, ce, ce, 0, 2 * ce, ce);
  }
  dconv(stream, cx, d_in, UP4, c4b, dt, y4, B, 128, H0, W0, 64);

  k_conv1x1<<<nblk((long)B * 2 * HW0), 256, 0, stream>>>(
      y4, (const float*)d_in[OUTC_W], (const float*)d_in[OUTC_B], OUT,
      B, 64, 2, (int)HW0);

  k_copy_chan<<<nblk(fst), 256, 0, stream>>>(test, OUT + fst, 1, fst, 0, 0, 0, 0);
}
